// fc_lstm_v0001_62311385530671
// MI455X (gfx1250) — matrix operand order repaired and checked by reading, not compiled, not run on hardware
//
#include <hip/hip_runtime.h>
#include <hip/hip_bf16.h>

#define HID   2048
#define GATE  (4 * HID)
#define BATCH 64
#define SEQ   16
#define FUT   16
#define INDIM 4096

typedef __attribute__((ext_vector_type(16))) __bf16 v16bf;
typedef __attribute__((ext_vector_type(8)))  float  v8f;

union FragU {
  v16bf v;
  uint4 q[2];
};

__device__ __forceinline__ unsigned short f2bf(float f) {
  unsigned int u = __float_as_uint(f);
  u += 0x7fffu + ((u >> 16) & 1u);   // round-to-nearest-even
  return (unsigned short)(u >> 16);
}

__device__ __forceinline__ float sigf(float x) {
  return 1.0f / (1.0f + __expf(-x));
}

// ---------------------------------------------------------------------------
// bf16 fragment loads (wave32); this code's contiguous k order, same for A and B.
// A (16x32 MxK): lanes 0-15 row M=lane, K=0..15; lanes 16-31 same rows,
// K=16..31 -> two 16B chunks at +0 / +8 halves.
__device__ __forceinline__ v16bf load_frag_a(const unsigned short* __restrict__ A,
                                             int lda, int kb, int lane) {
  int row  = lane & 15;
  int koff = kb + ((lane >> 4) << 4);
  const unsigned short* p = A + (long)row * lda + koff;
  FragU f;
  f.q[0] = *(const uint4*)(p);
  f.q[1] = *(const uint4*)(p + 8);
  return f.v;
}

// B (32x16 KxN): lane%16 = column N (= weight row; W is [N,K] row-major with K
// contiguous), lanes 0-15 K=kb..kb+15, lanes 16-31 K=kb+16..kb+31.
__device__ __forceinline__ v16bf load_frag_b(const unsigned short* __restrict__ W,
                                             int ldw, int row0, int kb, int lane) {
  int row  = row0 + (lane & 15);
  int koff = kb + ((lane >> 4) << 4);
  const unsigned short* p = W + (long)row * ldw + koff;
  FragU f;
  f.q[0] = *(const uint4*)(p);
  f.q[1] = *(const uint4*)(p + 8);
  return f.v;
}

#define WMMA_BF16(a, b, c) \
  __builtin_amdgcn_wmma_f32_16x16x32_bf16(false, (a), false, (b), (short)0, (c), false, false)

// ---------------------------------------------------------------------------
// Fused LSTM cell: gates = X@Wih^T + Hprev@Whh^T + b ; c' , h' computed
// in-register (no gates buffer).  Each wave owns one 16-wide j-tile and the
// four gate rows {j, j+H, j+2H, j+3H}, times two M-tiles sharing B fragments.
// Grid: x = HID/(16*8) = 16, y = 2 (m0 = y*16, second tile m0+32). Block 256.
// h_out must not alias Hprev (ping-pong buffers); c is updated in place.
__global__ __launch_bounds__(256)
void lstm_cell_wmma(const unsigned short* __restrict__ X,
                    const unsigned short* __restrict__ Wih,   // [4H, H]
                    const unsigned short* __restrict__ Hprev,
                    const unsigned short* __restrict__ Whh,   // [4H, H]
                    const float* __restrict__ bias,           // [4H]
                    float* __restrict__ c,                    // [64, H] in/out
                    unsigned short* __restrict__ h_out) {     // [64, H] bf16
  const int lane  = threadIdx.x & 31;
  const int wave  = threadIdx.x >> 5;
  const int jbase = (blockIdx.x * 8 + wave) * 16;
  const int m0    = blockIdx.y * 16;          // m-tiles: m0 and m0+32

  v8f acc[2][4];
#pragma unroll
  for (int mt = 0; mt < 2; ++mt)
#pragma unroll
    for (int t = 0; t < 4; ++t)
#pragma unroll
      for (int r = 0; r < 8; ++r) acc[mt][t][r] = 0.0f;

  const unsigned short* X0 = X + (long)m0 * HID;
  const unsigned short* X1 = X + (long)(m0 + 32) * HID;
  const unsigned short* H0 = Hprev + (long)m0 * HID;
  const unsigned short* H1 = Hprev + (long)(m0 + 32) * HID;

  for (int kb = 0; kb < HID; kb += 32) {
    v16bf ax0 = load_frag_a(X0, HID, kb, lane);
    v16bf ax1 = load_frag_a(X1, HID, kb, lane);
    v16bf ah0 = load_frag_a(H0, HID, kb, lane);
    v16bf ah1 = load_frag_a(H1, HID, kb, lane);
#pragma unroll
    for (int t = 0; t < 4; ++t) {
      v16bf bw = load_frag_b(Wih, HID, t * HID + jbase, kb, lane);
      acc[0][t] = WMMA_BF16(ax0, bw, acc[0][t]);
      acc[1][t] = WMMA_BF16(ax1, bw, acc[1][t]);
      v16bf bu = load_frag_b(Whh, HID, t * HID + jbase, kb, lane);
      acc[0][t] = WMMA_BF16(ah0, bu, acc[0][t]);
      acc[1][t] = WMMA_BF16(ah1, bu, acc[1][t]);
    }
  }

  // Epilogue: acc[mt][0..3][r] are (i,f,g,o) of element (m, j) with
  // m = m0 + mt*32 + 8*(lane>=16) + r, j = jbase + lane%16.
  const int j  = jbase + (lane & 15);
  const float bi = bias[j];
  const float bf = bias[HID + j];
  const float bg = bias[2 * HID + j];
  const float bo = bias[3 * HID + j];
#pragma unroll
  for (int mt = 0; mt < 2; ++mt) {
    const int mbase = m0 + mt * 32 + ((lane >> 4) << 3);
#pragma unroll
    for (int r = 0; r < 8; ++r) {
      const long idx = (long)(mbase + r) * HID + j;
      const float si = sigf(acc[mt][0][r] + bi);
      const float sf = sigf(acc[mt][1][r] + bf);
      const float gg = acc[mt][2][r] + bg;
      const float so = sigf(acc[mt][3][r] + bo);
      const float c2 = sf * c[idx] + si * tanhf(gg);
      const float h2 = so * tanhf(c2);
      c[idx] = c2;
      h_out[idx] = f2bf(h2);
    }
  }
}

// ---------------------------------------------------------------------------
// FC: out[64,N] = act(A[64,K] @ W[N,K]^T + bias). act: 1 = relu, 2 = sigmoid.
// Grid: x = N/512, y = 2 (two M-tiles per wave sharing B fragments). Block 256.
__global__ __launch_bounds__(256)
void fc_wmma(const unsigned short* __restrict__ A,
             const unsigned short* __restrict__ W, int K,
             const float* __restrict__ bias, int N,
             float* __restrict__ outF, unsigned short* __restrict__ outH,
             int act) {
  const int lane  = threadIdx.x & 31;
  const int wave  = threadIdx.x >> 5;
  const int nbase = (blockIdx.x * 8 + wave) * 64;
  const int m0    = blockIdx.y * 16;          // m-tiles: m0 and m0+32

  v8f acc[2][4];
#pragma unroll
  for (int mt = 0; mt < 2; ++mt)
#pragma unroll
    for (int t = 0; t < 4; ++t)
#pragma unroll
      for (int r = 0; r < 8; ++r) acc[mt][t][r] = 0.0f;

  const unsigned short* A0 = A + (long)m0 * K;
  const unsigned short* A1 = A + (long)(m0 + 32) * K;
  for (int kb = 0; kb < K; kb += 32) {
    v16bf a0 = load_frag_a(A0, K, kb, lane);
    v16bf a1 = load_frag_a(A1, K, kb, lane);
#pragma unroll
    for (int t = 0; t < 4; ++t) {
      v16bf b = load_frag_b(W, K, nbase + 16 * t, kb, lane);
      acc[0][t] = WMMA_BF16(a0, b, acc[0][t]);
      acc[1][t] = WMMA_BF16(a1, b, acc[1][t]);
    }
  }

#pragma unroll
  for (int mt = 0; mt < 2; ++mt) {
    const int mbase = m0 + mt * 32 + ((lane >> 4) << 3);
#pragma unroll
    for (int t = 0; t < 4; ++t) {
      const int n  = nbase + 16 * t + (lane & 15);
      const float bv = bias[n];
#pragma unroll
      for (int r = 0; r < 8; ++r) {
        float v = acc[mt][t][r] + bv;
        if (act == 1)      v = fmaxf(v, 0.0f);
        else if (act == 2) v = sigf(v);
        const long idx = (long)(mbase + r) * N + n;
        if (outF) outF[idx] = v;
        if (outH) outH[idx] = f2bf(v);
      }
    }
  }
}

// ---------------------------------------------------------------------------
__global__ void cvt_f32_bf16(const float* __restrict__ s,
                             unsigned short* __restrict__ d, long n) {
  long i = ((long)blockIdx.x * blockDim.x + threadIdx.x) * 4;
  const long stride = (long)gridDim.x * blockDim.x * 4;
  for (; i < n; i += stride) {
    float4 f = *(const float4*)(s + i);
    ushort4 o;
    o.x = f2bf(f.x); o.y = f2bf(f.y); o.z = f2bf(f.z); o.w = f2bf(f.w);
    *(ushort4*)(d + i) = o;
  }
}

__global__ void zero_u32(unsigned int* __restrict__ p, long n) {
  long i = (long)blockIdx.x * blockDim.x + threadIdx.x;
  const long stride = (long)gridDim.x * blockDim.x;
  for (; i < n; i += stride) p[i] = 0u;
}

__global__ void copy_u32(const unsigned int* __restrict__ s,
                         unsigned int* __restrict__ d, long n) {
  long i = (long)blockIdx.x * blockDim.x + threadIdx.x;
  const long stride = (long)gridDim.x * blockDim.x;
  for (; i < n; i += stride) d[i] = s[i];
}

// ---------------------------------------------------------------------------
extern "C" void kernel_launch(void* const* d_in, const int* in_sizes, int n_in,
                              void* d_out, int out_size, void* d_ws, size_t ws_size,
                              hipStream_t stream) {
  const float* x        = (const float*)d_in[0];
  const float* fc_en1_w = (const float*)d_in[1];
  const float* fc_en1_b = (const float*)d_in[2];
  const float* fc_en2_w = (const float*)d_in[3];
  const float* fc_en2_b = (const float*)d_in[4];
  const float* en_wih   = (const float*)d_in[5];
  const float* en_whh   = (const float*)d_in[6];
  const float* en_b     = (const float*)d_in[7];
  const float* de_wih   = (const float*)d_in[8];
  const float* de_whh   = (const float*)d_in[9];
  const float* de_b     = (const float*)d_in[10];
  const float* fc_de1_w = (const float*)d_in[11];
  const float* fc_de1_b = (const float*)d_in[12];
  const float* fc_de2_w = (const float*)d_in[13];
  const float* fc_de2_b = (const float*)d_in[14];
  float* out = (float*)d_out;
  (void)in_sizes; (void)n_in; (void)out_size; (void)ws_size;

  // ---- workspace carve-up (256B aligned) ----
  char* ws = (char*)d_ws;
  size_t off = 0;
  auto take = [&](size_t bytes) -> char* {
    char* p = ws + off;
    off += (bytes + 255) & ~(size_t)255;
    return p;
  };
  typedef unsigned short us;
  const long LSTM_W = (long)3 * GATE * HID;
  const long ST     = (long)BATCH * HID;       // one layer's h/c state elems

  us* en_wih_h = (us*)take((size_t)LSTM_W * 2);
  us* en_whh_h = (us*)take((size_t)LSTM_W * 2);
  us* de_wih_h = (us*)take((size_t)LSTM_W * 2);
  us* de_whh_h = (us*)take((size_t)LSTM_W * 2);
  us* fe1_h = (us*)take((size_t)1024 * INDIM * 2);
  us* fe2_h = (us*)take((size_t)HID * 1024 * 2);
  us* fd1_h = (us*)take((size_t)1024 * HID * 2);
  us* fd2_h = (us*)take((size_t)INDIM * 1024 * 2);
  us* x_h   = (us*)take((size_t)SEQ * BATCH * INDIM * 2);
  us* x1_h  = (us*)take((size_t)BATCH * 1024 * 2);
  us* x2_h  = (us*)take((size_t)BATCH * HID * 2);
  us* z1_h  = (us*)take((size_t)BATCH * 1024 * 2);
  us* he_buf[2] = { (us*)take((size_t)3 * ST * 2), (us*)take((size_t)3 * ST * 2) };
  us* hd_buf[2] = { (us*)take((size_t)3 * ST * 2), (us*)take((size_t)3 * ST * 2) };
  float* c_enc = (float*)take((size_t)3 * ST * 4);
  float* c_dec = (float*)take((size_t)3 * ST * 4);
  us* inp0 = (us*)take((size_t)ST * 2);

  // ---- one-time conversions (deterministic every call) ----
  auto cvt = [&](const float* s, us* d, long n) {
    cvt_f32_bf16<<<2048, 256, 0, stream>>>(s, d, n);
  };
  cvt(en_wih, en_wih_h, LSTM_W);
  cvt(en_whh, en_whh_h, LSTM_W);
  cvt(de_wih, de_wih_h, LSTM_W);
  cvt(de_whh, de_whh_h, LSTM_W);
  cvt(fc_en1_w, fe1_h, (long)1024 * INDIM);
  cvt(fc_en2_w, fe2_h, (long)HID * 1024);
  cvt(fc_de1_w, fd1_h, (long)1024 * HID);
  cvt(fc_de2_w, fd2_h, (long)INDIM * 1024);
  cvt(x, x_h, (long)SEQ * BATCH * INDIM);

  auto zero = [&](void* p, long words) {
    zero_u32<<<512, 256, 0, stream>>>((unsigned int*)p, words);
  };
  zero(he_buf[0], 3 * ST / 2);     // h_enc start state (bf16 = half words)
  zero(c_enc, 3 * ST);
  zero(hd_buf[0], 3 * ST / 2);     // layers 1,2 stay zero; layer 0 copied below
  zero(c_dec, 3 * ST);
  zero(inp0, ST / 2);

  auto fc = [&](const us* A, const us* W, int K, const float* bias, int N,
                float* oF, us* oH, int act) {
    fc_wmma<<<dim3(N / 512, 2), 256, 0, stream>>>(A, W, K, bias, N, oF, oH, act);
  };
  const dim3 lstm_grid(HID / 128, 2);
  auto lstm = [&](const us* X, const us* Wih, const us* Hp, const us* Whh,
                  const float* bias, float* c, us* ho) {
    lstm_cell_wmma<<<lstm_grid, 256, 0, stream>>>(X, Wih, Hp, Whh, bias, c, ho);
  };

  // ---- encoder (h ping-pong: read he_buf[t&1], write he_buf[1-(t&1)]) ----
  for (int t = 0; t < SEQ; ++t) {
    us* hr = he_buf[t & 1];
    us* hw = he_buf[1 - (t & 1)];
    const us* xt = x_h + (long)t * BATCH * INDIM;
    fc(xt, fe1_h, INDIM, fc_en1_b, 1024, nullptr, x1_h, 1);
    fc(x1_h, fe2_h, 1024, fc_en2_b, HID, nullptr, x2_h, 1);
    for (int l = 0; l < 3; ++l) {
      const us* inp = (l == 0) ? x2_h : (hw + (long)(l - 1) * ST);
      lstm(inp, en_wih_h + (long)l * GATE * HID, hr + (long)l * ST,
           en_whh_h + (long)l * GATE * HID, en_b + (long)l * GATE,
           c_enc + (long)l * ST, hw + (long)l * ST);
    }
  }
  // SEQ=16 (even): final encoder state lives in he_buf[0]; h_e3 = layer 2.

  // ---- decoder init: h_d0 = h_e3, everything else zero ----
  copy_u32<<<256, 256, 0, stream>>>(
      (const unsigned int*)(he_buf[0] + (long)2 * ST),
      (unsigned int*)hd_buf[0], ST / 2);

  // ---- decoder ----
  for (int t = 0; t < FUT; ++t) {
    us* hr = hd_buf[t & 1];
    us* hw = hd_buf[1 - (t & 1)];
    for (int l = 0; l < 3; ++l) {
      // l==0 input: zeros at t==0, else previous step's h_d3 (in read buffer).
      const us* inp = (l == 0) ? ((t == 0) ? inp0 : (hr + (long)2 * ST))
                               : (hw + (long)(l - 1) * ST);
      lstm(inp, de_wih_h + (long)l * GATE * HID, hr + (long)l * ST,
           de_whh_h + (long)l * GATE * HID, de_b + (long)l * GATE,
           c_dec + (long)l * ST, hw + (long)l * ST);
    }
    fc(hw + (long)2 * ST, fd1_h, HID, fc_de1_b, 1024, nullptr, z1_h, 1);
    fc(z1_h, fd2_h, 1024, fc_de2_b, INDIM,
       out + (long)t * BATCH * INDIM, nullptr, 2);
  }
}
